// RotaryAttention_3298534883446
// MI455X (gfx1250) — matrix operand order repaired and checked by reading, not compiled, not run on hardware
//
#include <hip/hip_runtime.h>
#include <stdint.h>

// Rotary attention, MI455X (gfx1250): bf16 WMMA, TDM tensor-DMA GEMM staging,
// async-to-LDS attention staging. b=16 s=512 d=2048 H=16 hd=128

#define B_  16
#define S_  512
#define D_  2048
#define H_  16
#define HD_ 128
#define M_  (B_*S_)

typedef __attribute__((ext_vector_type(16))) __bf16 bf16x16;
typedef __attribute__((ext_vector_type(8)))  float  f32x8;
typedef int v4i_ __attribute__((vector_size(16)));   // matches async builtin param

union Frag { bf16x16 v; uint32_t u[8]; };

__device__ __forceinline__ uint16_t f2bf(float f){
  union { float f; uint32_t u; } c; c.f = f;
  uint32_t r = c.u + 0x7FFFu + ((c.u >> 16) & 1u);   // round-to-nearest-even
  return (uint16_t)(r >> 16);
}
__device__ __forceinline__ float bf2f(uint16_t h){
  union { uint32_t u; float f; } c; c.u = ((uint32_t)h) << 16;
  return c.f;
}
// A-operand (16x32 bf16) per-lane K pattern: vgpr j, lane-half h
__device__ __forceinline__ int kpatA(int j, int half){
  return (j < 4 ? 2*j : 16 + 2*(j-4)) + 8*half;
}
// B-operand (32x16 bf16) per-lane K pattern
__device__ __forceinline__ int kpatB(int j, int half){
  return (j < 4 ? 2*j : 16 + 2*(j-4)) + 8*half;
}

// ---- CDNA5 async global->LDS copy (ASYNCcnt-tracked) ----
#if __has_builtin(__builtin_amdgcn_global_load_async_to_lds_b128)
__device__ __forceinline__ void async_b128(const uint16_t* g, uint16_t* l){
  __builtin_amdgcn_global_load_async_to_lds_b128(
      (__attribute__((address_space(1))) v4i_*)g,
      (__attribute__((address_space(3))) v4i_*)l, 0, 0);
}
#else
__device__ __forceinline__ void async_b128(const uint16_t* g, uint16_t* l){
  uint32_t loff = (uint32_t)(uintptr_t)(__attribute__((address_space(3))) uint16_t*)l;
  uint64_t ga   = (uint64_t)(uintptr_t)g;
  asm volatile("global_load_async_to_lds_b128 %0, %1, off"
               :: "v"(loff), "v"(ga) : "memory");
}
#endif

#if __has_builtin(__builtin_amdgcn_s_wait_asynccnt)
#define WAIT_ASYNC(n) __builtin_amdgcn_s_wait_asynccnt(n)
#else
#define WAIT_ASYNC(n) asm volatile("s_wait_asynccnt " #n ::: "memory")
#endif

// ---- CDNA5 Tensor Data Mover: 2D tile DMA global->LDS (TENSORcnt-tracked) ----
#if __has_builtin(__builtin_amdgcn_tensor_load_to_lds)
#define HAVE_TDM 1
typedef unsigned int u32x4 __attribute__((ext_vector_type(4)));
typedef int i32x8 __attribute__((ext_vector_type(8)));
typedef int i32x4 __attribute__((ext_vector_type(4)));
// D# per CDNA5 ISA: 2D tensor, bf16 (data_size=2B),
// tile_w x tile_h tile at gbase, row stride in elements; groups 2/3 zero (<=2D).
__device__ __forceinline__ void tdm_load_2d(const uint16_t* gbase, uint32_t lds_off,
                                            int tensor_w, int tensor_h,
                                            int tile_w, int tile_h, int row_stride){
  uint64_t ga = (uint64_t)(uintptr_t)gbase;
  u32x4 g0;
  g0.x = 1u;                                             // count=1, user D#
  g0.y = lds_off;                                        // lds_addr (bytes)
  g0.z = (uint32_t)ga;                                   // global_addr[31:0]
  g0.w = (uint32_t)((ga >> 32) & 0x01FFFFFFu) | (2u << 30); // addr[56:32] | type=2
  i32x8 g1;
  g1[0] = 1 << 16;                                       // data_size=1 (2 bytes)
  g1[1] = (tensor_w & 0xFFFF) << 16;                     // tensor_dim0[15:0]
  g1[2] = ((tensor_w >> 16) & 0xFFFF) | ((tensor_h & 0xFFFF) << 16); // dim0 hi | dim1 lo
  g1[3] = ((tensor_h >> 16) & 0xFFFF) | ((tile_w & 0xFFFF) << 16);   // dim1 hi | tile_dim0
  g1[4] = tile_h & 0xFFFF;                               // tile_dim1, tile_dim2=0
  g1[5] = row_stride;                                    // tensor_dim0_stride[31:0]
  g1[6] = 0;                                             // stride hi | dim1_stride lo
  g1[7] = 0;
  i32x4 gz4 = {0,0,0,0};
  i32x8 gz8 = {0,0,0,0,0,0,0,0};
  __builtin_amdgcn_tensor_load_to_lds(g0, g1, gz4, gz4, gz8, 0);
}
#define WAIT_TENSOR(n) __builtin_amdgcn_s_wait_tensorcnt(n)
#else
#define HAVE_TDM 0
#endif

__device__ __forceinline__ uint32_t lds_off_u32(const void* p){
  return (uint32_t)(uintptr_t)(__attribute__((address_space(3))) const void*)p;
}

// ---------------- conversion kernels ----------------
__global__ void k_cvt_w(const float* __restrict__ w, uint16_t* __restrict__ wt){
  int idx = blockIdx.x*256 + threadIdx.x;      // idx = n*D + k
  int n = idx / D_, k = idx - n*D_;
  wt[idx] = f2bf(w[(size_t)k*D_ + n]);         // store W^T [N,K]
}
__global__ void k_cvt_x(const float* __restrict__ x, uint16_t* __restrict__ xb){
  int idx = blockIdx.x*256 + threadIdx.x;
  xb[idx] = f2bf(x[idx]);
}

// ---------------- RoPE + head reshape ----------------
// mode 0: rope, out[b][h][s][hd]   (Q,K)
// mode 1: plain, out[b][h][hd][s]  (V transposed -> contiguous B-frag loads)
__global__ void k_rope_reshape(const uint16_t* __restrict__ in,
                               uint16_t* __restrict__ out, int mode){
  int idx = blockIdx.x*256 + threadIdx.x;      // B*S*H*64 pairs
  int p  = idx & 63;
  int h  = (idx >> 6) & (H_-1);
  int bs = idx >> 10;
  int s  = bs & (S_-1);
  int b  = bs >> 9;
  size_t ib = (size_t)bs*D_ + h*HD_ + 2*p;
  float e0 = bf2f(in[ib]), e1 = bf2f(in[ib+1]);
  if (mode == 0){
    float freq = (float)s * __powf(10000.0f, -(float)p * (1.0f/64.0f));
    float si, c; __sincosf(freq, &si, &c);
    size_t o = ((size_t)(b*H_ + h)*S_ + s)*HD_ + 2*p;
    out[o]   = f2bf(e0*c - e1*si);
    out[o+1] = f2bf(e1*c + e0*si);
  } else {
    size_t o = ((size_t)(b*H_ + h)*HD_ + 2*p)*S_ + s;
    out[o]      = f2bf(e0);
    out[o + S_] = f2bf(e1);
  }
}

// ---------------- GEMM: C[M,N] = A[M,K] * Bt[N,K]^T + bias ----------------
// 256 thr (8 waves), block tile 64x128, wave tile 32x32, K-step 64.
// Tile staging: TDM tensor DMA (wave 0 issues; TENSORcnt), double-buffered.
template <typename OutT>
__global__ __launch_bounds__(256) void k_gemm(
    const uint16_t* __restrict__ A, const uint16_t* __restrict__ Bt,
    const float* __restrict__ bias, OutT* __restrict__ C,
    int M, int N, int K)
{
  __shared__ uint16_t lA[2][64][64];     // 16 KB
  __shared__ uint16_t lB[2][128][64];    // 32 KB
  const int tid = threadIdx.x, lane = tid & 31, wid = tid >> 5;
  const int half = lane >> 4, l16 = lane & 15;
  const int m0 = blockIdx.y * 64, n0 = blockIdx.x * 128;
  const int wm = wid >> 2, wn = wid & 3;

  f32x8 zf = {0.f,0.f,0.f,0.f,0.f,0.f,0.f,0.f};
  f32x8 acc[2][2];
  acc[0][0]=zf; acc[0][1]=zf; acc[1][0]=zf; acc[1][1]=zf;

#if HAVE_TDM
  auto stage = [&](int k0, int buf){     // 2 TDM descriptors: A tile + B tile
    tdm_load_2d(&A[(size_t)m0*K + k0],  lds_off_u32(&lA[buf][0][0]), K, M,  64,  64, K);
    tdm_load_2d(&Bt[(size_t)n0*K + k0], lds_off_u32(&lB[buf][0][0]), K, N,  64, 128, K);
  };
  if (wid == 0) stage(0, 0);
#else
  auto stage = [&](int k0, int buf){
    #pragma unroll
    for (int i=0;i<2;i++){               // A: 64x64 = 512 x b128
      int ch = tid + 256*i;
      int r = ch >> 3, c8 = (ch & 7) * 8;
      async_b128(&A[(size_t)(m0+r)*K + k0 + c8], &lA[buf][r][c8]);
    }
    #pragma unroll
    for (int i=0;i<4;i++){               // B: 128x64 = 1024 x b128
      int ch = tid + 256*i;
      int r = ch >> 3, c8 = (ch & 7) * 8;
      async_b128(&Bt[(size_t)(n0+r)*K + k0 + c8], &lB[buf][r][c8]);
    }
  };
  stage(0, 0);
#endif

  int buf = 0;
  for (int k0 = 0; k0 < K; k0 += 64, buf ^= 1) {
#if HAVE_TDM
    if (wid == 0){
      if (k0 + 64 < K) { stage(k0 + 64, buf ^ 1); WAIT_TENSOR(2); }
      else             { WAIT_TENSOR(0); }
    }
#else
    if (k0 + 64 < K) { stage(k0 + 64, buf ^ 1); WAIT_ASYNC(6); }
    else             { WAIT_ASYNC(0); }
#endif
    __syncthreads();
    #pragma unroll
    for (int kh=0; kh<2; kh++){
      Frag a[2], b[2];
      #pragma unroll
      for (int i=0;i<2;i++){
        int row = wm*32 + i*16 + l16;
        #pragma unroll
        for (int j=0;j<8;j++)
          a[i].u[j] = *(const uint32_t*)&lA[buf][row][kh*32 + kpatA(j,half)];
      }
      #pragma unroll
      for (int t=0;t<2;t++){
        int row = wn*32 + t*16 + l16;
        #pragma unroll
        for (int j=0;j<8;j++)
          b[t].u[j] = *(const uint32_t*)&lB[buf][row][kh*32 + kpatB(j,half)];
      }
      #pragma unroll
      for (int i=0;i<2;i++)
        #pragma unroll
        for (int t=0;t<2;t++)
          acc[i][t] = __builtin_amdgcn_wmma_f32_16x16x32_bf16(
              false, a[i].v, false, b[t].v, (short)0, acc[i][t], false, false);
    }
    __syncthreads();
  }
  #pragma unroll
  for (int i=0;i<2;i++){
    #pragma unroll
    for (int t=0;t<2;t++){
      int col = n0 + wn*32 + t*16 + l16;
      float bb = bias[col];
      #pragma unroll
      for (int r=0;r<8;r++){
        int row = m0 + wm*32 + i*16 + r + half*8;
        float v = acc[i][t][r] + bb;
        if constexpr (sizeof(OutT) == 2) C[(size_t)row*N + col] = f2bf(v);
        else                             C[(size_t)row*N + col] = v;
      }
    }
  }
}

// ---------------- flash attention (causal) ----------------
// grid = B*H*(S/128); block = 8 waves; wave = 16 query rows.
// K/V tiles async-staged to LDS once per block, double-buffered;
// waves past causal bound skip compute but keep barriers.
__global__ __launch_bounds__(256) void k_attn(
    const uint16_t* __restrict__ Q, const uint16_t* __restrict__ Kh,
    const uint16_t* __restrict__ Vt, uint16_t* __restrict__ O)
{
  __shared__ uint16_t lK[2][32][128];    // 16 KB
  __shared__ uint16_t lV[2][128][32];    // 16 KB
  __shared__ uint16_t lP[8][16][32];     //  8 KB per-wave P staging
  const int tid = threadIdx.x, lane = tid & 31, wid = tid >> 5;
  const int half = lane >> 4, l16 = lane & 15;
  const int qt = blockIdx.x & 3;                  // S/128 = 4 q-tiles
  const int bh = blockIdx.x >> 2;                 // b*H + h
  const uint16_t* q  = Q  + (size_t)bh * S_ * HD_;
  const uint16_t* kk = Kh + (size_t)bh * S_ * HD_;
  const uint16_t* vt = Vt + (size_t)bh * HD_ * S_;   // [HD][S]
  const int mb = qt*128 + wid*16;                 // query row base
  const float scale = 0.08838834764831845f;       // 1/sqrt(128)

  Frag aQ[4];                                     // Q rows, 4 hd-chunks of 32
  {
    int row = mb + l16;
    #pragma unroll
    for (int c=0;c<4;c++)
      #pragma unroll
      for (int j=0;j<8;j++)
        aQ[c].u[j] = *(const uint32_t*)&q[(size_t)row*HD_ + c*32 + kpatA(j,half)];
  }

  f32x8 zf = {0.f,0.f,0.f,0.f,0.f,0.f,0.f,0.f};
  f32x8 o[8];
  float mst[8], lst[8];
  #pragma unroll
  for (int n=0;n<8;n++) o[n] = zf;
  #pragma unroll
  for (int r=0;r<8;r++){ mst[r] = -1e30f; lst[r] = 0.f; }

  auto stage_kv = [&](int j, int buf){
    int sk0 = j*32;
    #pragma unroll
    for (int i=0;i<2;i++){
      int ch = tid + 256*i;
      int kr = ch >> 4, kc = (ch & 15) * 8;       // K tile 32x128
      async_b128(&kk[(size_t)(sk0+kr)*HD_ + kc], &lK[buf][kr][kc]);
      int vr = ch >> 2, vc = (ch & 3) * 8;        // V^T tile 128x32
      async_b128(&vt[(size_t)vr*S_ + sk0 + vc], &lV[buf][vr][vc]);
    }
  };

  const int jt_wave = (mb + 15) >> 5;             // this wave's causal bound
  const int jt_blk  = (qt*128 + 127) >> 5;        // block-uniform bound
  stage_kv(0, 0);
  int buf = 0;
  for (int jv = 0; jv <= jt_blk; jv++, buf ^= 1){
    if (jv < jt_blk) { stage_kv(jv+1, buf^1); WAIT_ASYNC(4); }
    else             { WAIT_ASYNC(0); }
    __syncthreads();
    if (jv <= jt_wave){
      const int sk0 = jv*32;
      f32x8 sacc[2]; sacc[0]=zf; sacc[1]=zf;
      #pragma unroll
      for (int c=0;c<4;c++){
        #pragma unroll
        for (int t=0;t<2;t++){
          Frag bk;
          #pragma unroll
          for (int j=0;j<8;j++)
            bk.u[j] = *(const uint32_t*)&lK[buf][t*16 + l16][c*32 + kpatB(j,half)];
          sacc[t] = __builtin_amdgcn_wmma_f32_16x16x32_bf16(
              false, aQ[c].v, false, bk.v, (short)0, sacc[t], false, false);
        }
      }
      // causal mask + online softmax (row r lives in one 16-lane half)
      float p[2][8];
      #pragma unroll
      for (int r=0;r<8;r++){
        int mrow = mb + r + half*8;
        float v0 = sacc[0][r]*scale, v1 = sacc[1][r]*scale;
        if (sk0 + l16      > mrow) v0 = -1e9f;
        if (sk0 + 16 + l16 > mrow) v1 = -1e9f;
        float vm = fmaxf(v0, v1);
        #pragma unroll
        for (int off=1; off<16; off<<=1)
          vm = fmaxf(vm, __shfl_xor(vm, off, 32));
        float mnew  = fmaxf(mst[r], vm);
        float alpha = __expf(mst[r] - mnew);
        float p0 = __expf(v0 - mnew), p1 = __expf(v1 - mnew);
        p[0][r] = p0; p[1][r] = p1;
        float rs = p0 + p1;
        #pragma unroll
        for (int off=1; off<16; off<<=1)
          rs += __shfl_xor(rs, off, 32);
        lst[r] = lst[r]*alpha + rs;
        mst[r] = mnew;
        #pragma unroll
        for (int n=0;n<8;n++) o[n][r] *= alpha;
      }
      // stage P (C layout -> row-major) and reload as A fragment
      #pragma unroll
      for (int t=0;t<2;t++)
        #pragma unroll
        for (int r=0;r<8;r++)
          lP[wid][r + half*8][t*16 + l16] = f2bf(p[t][r]);
      asm volatile("s_wait_dscnt 0" ::: "memory");  // cross-lane LDS visibility
      Frag aP;
      #pragma unroll
      for (int j=0;j<8;j++)
        aP.u[j] = *(const uint32_t*)&lP[wid][l16][kpatA(j,half)];
      // O += P @ V
      #pragma unroll
      for (int n=0;n<8;n++){
        Frag bv;
        #pragma unroll
        for (int j=0;j<8;j++)
          bv.u[j] = *(const uint32_t*)&lV[buf][n*16 + l16][kpatB(j,half)];
        o[n] = __builtin_amdgcn_wmma_f32_16x16x32_bf16(
            false, aP.v, false, bv.v, (short)0, o[n], false, false);
      }
    }
    __syncthreads();
  }
  // epilogue: normalize, write attn_out bf16 [b*s, h*HD+hd]
  const int b = bh >> 4, h = bh & 15;
  #pragma unroll
  for (int n=0;n<8;n++){
    #pragma unroll
    for (int r=0;r<8;r++){
      int srow = mb + r + half*8;
      float val = o[n][r] / lst[r];
      size_t oidx = ((size_t)(b*S_ + srow))*D_ + h*HD_ + n*16 + l16;
      O[oidx] = f2bf(val);
    }
  }
}

// ---------------- launcher ----------------
extern "C" void kernel_launch(void* const* d_in, const int* in_sizes, int n_in,
                              void* d_out, int out_size, void* d_ws, size_t ws_size,
                              hipStream_t stream)
{
  const float* x  = (const float*)d_in[0];
  // d_in[1] = mask (causal, handled analytically)
  const float* wq = (const float*)d_in[2];
  const float* bq = (const float*)d_in[3];
  const float* wk = (const float*)d_in[4];
  const float* bk = (const float*)d_in[5];
  const float* wv = (const float*)d_in[6];
  const float* bv = (const float*)d_in[7];
  const float* wo = (const float*)d_in[8];
  const float* bo = (const float*)d_in[9];
  float* out = (float*)d_out;

  char* ws = (char*)d_ws;
  const size_t WSZ = (size_t)D_*D_*2;   // 8 MB per weight (bf16)
  const size_t XSZ = (size_t)M_*D_*2;   // 32 MB per activation (bf16)
  uint16_t* wqT  = (uint16_t*)(ws);
  uint16_t* wkT  = (uint16_t*)(ws + WSZ);
  uint16_t* wvT  = (uint16_t*)(ws + 2*WSZ);
  uint16_t* woT  = (uint16_t*)(ws + 3*WSZ);
  uint16_t* xb   = (uint16_t*)(ws + 4*WSZ);
  uint16_t* tmp  = (uint16_t*)(ws + 4*WSZ + XSZ);
  uint16_t* Qh   = (uint16_t*)(ws + 4*WSZ + 2*XSZ);
  uint16_t* Kht  = (uint16_t*)(ws + 4*WSZ + 3*XSZ);
  uint16_t* Vt   = (uint16_t*)(ws + 4*WSZ + 4*XSZ);
  uint16_t* attn = (uint16_t*)(ws + 4*WSZ + 5*XSZ);

  dim3 blk(256);
  k_cvt_w<<<D_*D_/256, blk, 0, stream>>>(wq, wqT);
  k_cvt_w<<<D_*D_/256, blk, 0, stream>>>(wk, wkT);
  k_cvt_w<<<D_*D_/256, blk, 0, stream>>>(wv, wvT);
  k_cvt_w<<<D_*D_/256, blk, 0, stream>>>(wo, woT);
  k_cvt_x<<<M_*D_/256, blk, 0, stream>>>(x, xb);

  dim3 ggrid(D_/128, M_/64);
  const int rgrid = (B_*S_*H_*64)/256;
  k_gemm<uint16_t><<<ggrid, blk, 0, stream>>>(xb, wqT, bq, tmp, M_, D_, D_);
  k_rope_reshape<<<rgrid, blk, 0, stream>>>(tmp, Qh, 0);
  k_gemm<uint16_t><<<ggrid, blk, 0, stream>>>(xb, wkT, bk, tmp, M_, D_, D_);
  k_rope_reshape<<<rgrid, blk, 0, stream>>>(tmp, Kht, 0);
  k_gemm<uint16_t><<<ggrid, blk, 0, stream>>>(xb, wvT, bv, tmp, M_, D_, D_);
  k_rope_reshape<<<rgrid, blk, 0, stream>>>(tmp, Vt, 1);

  k_attn<<<B_*H_*(S_/128), blk, 0, stream>>>(Qh, Kht, Vt, attn);

  k_gemm<float><<<ggrid, blk, 0, stream>>>(attn, woT, bo, out, M_, D_, D_);
}
